// MultiHeadAttention_86732569575441
// MI455X (gfx1250) — compile-verified, matrix operand order checked by reading, not run on hardware
//
#include <hip/hip_runtime.h>
#include <hip/hip_bf16.h>

// ---------------------------------------------------------------------------
// Multi-head attention, CDNA5 (gfx1250), wave32, WMMA f16 (f32 accumulate),
// TDM (tensor_load_to_lds) double-buffered LDS staging for the GEMMs.
// B=2, N=2048, DIM=1024, H=16, D=64, SCALE=1/8.
// ---------------------------------------------------------------------------

typedef __attribute__((ext_vector_type(16))) _Float16 v16h;
typedef __attribute__((ext_vector_type(8)))  float    v8f;
typedef __attribute__((ext_vector_type(4)))  unsigned int v4u;
typedef __attribute__((ext_vector_type(8)))  int      v8i;
typedef __attribute__((ext_vector_type(4)))  int      v4i;

#define DIMC   1024
#define SEQN   2048
#define NHEAD  16
#define HDIM   64
#define NBATCH 2
#define ATTN_SCALE 0.125f

__device__ __forceinline__ v8f wmma16(v16h a, v16h b, v8f c) {
    // 8 args: (neg_a, A, neg_b, B, c_mod, C, reuse_a, reuse_b)
    return __builtin_amdgcn_wmma_f32_16x16x32_f16(
        false, a, false, b, (short)0, c, false, false);
}

// LDS byte offset of a __shared__ object (generic ptr low 32 bits = LDS addr)
__device__ __forceinline__ unsigned lds_off(const void* p) {
    return (unsigned)(unsigned long long)(uintptr_t)p;
}

// ---------------------------------------------------------------------------
// TDM: load a 2D tile (tile_d0 x tile_d1 elements, f16) from a row-major
// tensor (tensor_d0 cols, stride0 elements between rows) into LDS (packed,
// row-major tile). One DMA op per issuing wave; tracked by TENSORcnt.
// Descriptor layout per CDNA5 ISA ch.8 (D# groups 0..3).
// ---------------------------------------------------------------------------
__device__ __forceinline__ void tdm_load_2d_f16(unsigned lds_addr,
                                                const void* gaddr,
                                                unsigned tensor_d0,
                                                unsigned tensor_d1,
                                                unsigned tile_d0,
                                                unsigned tile_d1,
                                                unsigned stride0) {
    unsigned long long ga = (unsigned long long)(uintptr_t)gaddr;
    v4u g0;
    g0[0] = 1u;                                           // count=1 (valid), user
    g0[1] = lds_addr;                                     // lds_addr [63:32]
    g0[2] = (unsigned)(ga & 0xFFFFFFFFu);                 // global_addr[31:0]
    g0[3] = (unsigned)((ga >> 32) & 0x01FFFFFFu)          // global_addr[56:32]
            | (2u << 30);                                 // type=2 ("image")
    v8i g1;
    g1[0] = (int)(1u << 16);                              // data_size=1 (2B), wg_mask=0
    g1[1] = (int)((tensor_d0 & 0xFFFFu) << 16);           // tensor_dim0[15:0]
    g1[2] = (int)(((tensor_d0 >> 16) & 0xFFFFu)           // tensor_dim0[31:16]
            | ((tensor_d1 & 0xFFFFu) << 16));             // tensor_dim1[15:0]
    g1[3] = (int)(((tensor_d1 >> 16) & 0xFFFFu)           // tensor_dim1[31:16]
            | ((tile_d0 & 0xFFFFu) << 16));               // tile_dim0
    g1[4] = (int)(tile_d1 & 0xFFFFu);                     // tile_dim1 (tile_dim2=0)
    g1[5] = (int)stride0;                                 // tensor_dim0_stride[31:0]
    g1[6] = 0;                                            // stride0 hi / stride1 lo
    g1[7] = 0;
    v4i z4 = {0, 0, 0, 0};                                // groups 2/3: unused (2D)
    v8i z8 = {0, 0, 0, 0, 0, 0, 0, 0};                    // trailing group (clang-23 form)
    __builtin_amdgcn_tensor_load_to_lds(g0, g1, z4, z4, z8, 0);
}

// ---------------------------------------------------------------------------
// fp32 -> f16 conversion (vector-4)
// ---------------------------------------------------------------------------
__global__ void cvt_f32_f16_kernel(const float* __restrict__ src,
                                   _Float16* __restrict__ dst, int n4) {
    int i = blockIdx.x * blockDim.x + threadIdx.x;
    if (i < n4) {
        float4 v = ((const float4*)src)[i];
        dst[i*4 + 0] = (_Float16)v.x; dst[i*4 + 1] = (_Float16)v.y;
        dst[i*4 + 2] = (_Float16)v.z; dst[i*4 + 3] = (_Float16)v.w;
    }
}

// ---------------------------------------------------------------------------
// GEMM: C[m,c] = sum_k A[m,k] * W[c,k]    (A: 4096x1024 f16, W: 1024x1024 f16)
// Block = 256 threads = 8 waves (2 M x 4 N), block tile 64x128,
// wave tile 32x32 (2x2 WMMA fragments), K-step 64 (8 WMMAs per sync period).
// A 64x64 tile and W 128x64 tile staged in LDS by the Tensor Data Mover,
// double-buffered: wave 0 issues TDM for step i+1 while all waves run WMMA
// on step i out of LDS. Sync: s_wait_tensorcnt 0 + workgroup barrier.
// MODE 0: store f16 at (B,H,N,D)       [Q and K]
// MODE 1: store f16 at (B,H,D,N)       [V transposed]
// MODE 2: store fp32 + bias row-major  [final projection]
// ---------------------------------------------------------------------------
template <int MODE>
__global__ void gemm_wmma_kernel(const _Float16* __restrict__ A,
                                 const _Float16* __restrict__ W,
                                 _Float16* __restrict__ outH,
                                 float* __restrict__ outF,
                                 const float* __restrict__ bias) {
    const int K = DIMC;
    const int KSTEP = 64;
    __shared__ _Float16 Abuf[2][64 * KSTEP];    // 8 KB each
    __shared__ _Float16 Bbuf[2][128 * KSTEP];   // 16 KB each

    const int tid  = threadIdx.x;
    const int lane = tid & 31;
    const int w    = tid >> 5;                 // wave 0..7
    const int bm   = blockIdx.x >> 3;          // 0..63
    const int bn   = blockIdx.x & 7;           // 0..7
    const int Mblk  = bm * 64;
    const int Nblk  = bn * 128;
    const int wmOff = (w >> 2) * 32;           // wave M offset in block tile
    const int wnOff = (w & 3) * 32;            // wave N offset in block tile
    const int lm    = lane & 15;
    const int koff  = (lane >> 4) * 16;        // fragment K-half select

    const int NSTEP = K / KSTEP;               // 16 k-steps

    // Prologue: wave 0 issues TDM for k-step 0 into buffer 0
    if (w == 0) {
        tdm_load_2d_f16(lds_off(&Abuf[0][0]), A + (size_t)Mblk * K,
                        K, 4096, KSTEP, 64, K);
        tdm_load_2d_f16(lds_off(&Bbuf[0][0]), W + (size_t)Nblk * K,
                        K, DIMC, KSTEP, 128, K);
    }

    v8f acc00 = {}, acc01 = {}, acc10 = {}, acc11 = {};
    for (int i = 0; i < NSTEP; ++i) {
        if (w == 0) __builtin_amdgcn_s_wait_tensorcnt(0);
        __syncthreads();                        // buf[i&1] ready for all waves
        if (w == 0 && (i + 1) < NSTEP) {        // overlap DMA of step i+1
            int k1 = (i + 1) * KSTEP;
            tdm_load_2d_f16(lds_off(&Abuf[(i + 1) & 1][0]),
                            A + (size_t)Mblk * K + k1, K, 4096, KSTEP, 64, K);
            tdm_load_2d_f16(lds_off(&Bbuf[(i + 1) & 1][0]),
                            W + (size_t)Nblk * K + k1, K, DIMC, KSTEP, 128, K);
        }
        const _Float16* At = &Abuf[i & 1][0] + koff;
        const _Float16* Bt = &Bbuf[i & 1][0] + koff;
        #pragma unroll
        for (int kd = 0; kd < KSTEP; kd += 32) {
            v16h a0 = *(const v16h*)(At + (wmOff + lm) * KSTEP + kd);
            v16h a1 = *(const v16h*)(At + (wmOff + 16 + lm) * KSTEP + kd);
            v16h b0 = *(const v16h*)(Bt + (wnOff + lm) * KSTEP + kd);
            v16h b1 = *(const v16h*)(Bt + (wnOff + 16 + lm) * KSTEP + kd);
            acc00 = wmma16(a0, b0, acc00);
            acc01 = wmma16(a0, b1, acc01);
            acc10 = wmma16(a1, b0, acc10);
            acc11 = wmma16(a1, b1, acc11);
        }
    }

    const int rquad = (lane >> 4) * 8;         // C rows: r + 8*(lane>=16)
    v8f accs[2][2] = {{acc00, acc01}, {acc10, acc11}};
    for (int mt = 0; mt < 2; ++mt) {
        for (int nt = 0; nt < 2; ++nt) {
            for (int r = 0; r < 8; ++r) {
                int m = Mblk + wmOff + mt * 16 + r + rquad;  // token row
                int c = Nblk + wnOff + nt * 16 + lm;         // feature col
                float vacc = accs[mt][nt][r];
                if (MODE == 2) {
                    outF[(size_t)m * DIMC + c] = vacc + bias[c];
                } else {
                    int b = m >> 11, n = m & (SEQN - 1);
                    int h = c >> 6, d = c & (HDIM - 1);
                    size_t off;
                    if (MODE == 0)       // (B,H,N,D)
                        off = ((size_t)(b * NHEAD + h) * SEQN + n) * HDIM + d;
                    else                 // (B,H,D,N)  -- transposed V
                        off = ((size_t)(b * NHEAD + h) * HDIM + d) * SEQN + n;
                    outH[off] = (_Float16)vacc;
                }
            }
        }
    }
}

// ---------------------------------------------------------------------------
// Attention: one workgroup per (b*h, 32-query block). 256 threads = 8 waves.
// Full 32x2048 fp32 score strip resident in LDS (CDNA5 320KB LDS) with a
// 4-float row skew; single-pass softmax; WMMA for Q*Kt and P*V.
// ---------------------------------------------------------------------------
#define SROW 2052   // padded row stride in floats (2052 % 64 == 4)

__global__ void attn_kernel(const _Float16* __restrict__ qh,
                            const _Float16* __restrict__ kh,
                            const _Float16* __restrict__ vT,
                            _Float16* __restrict__ oh) {
    extern __shared__ char smem[];
    float* S     = (float*)smem;                         // [32][SROW]
    float* red   = (float*)(smem + 32 * SROW * 4);       // [32][8]
    float* rstat = red + 32 * 8;                         // [32] max, then 1/sum

    const int tid  = threadIdx.x;
    const int lane = tid & 31;
    const int w    = tid >> 5;
    const int bh   = blockIdx.x >> 6;                    // 0..31
    const int nblk = blockIdx.x & 63;                    // 32-row query block
    const int lm   = lane & 15;
    const int koff = (lane >> 4) * 16;

    const _Float16* Qb = qh + (size_t)bh * SEQN * HDIM;
    const _Float16* Kb = kh + (size_t)bh * SEQN * HDIM;
    const _Float16* Vb = vT + (size_t)bh * HDIM * SEQN;

    // ---- Phase 1: S = SCALE * Q K^T -------------------------------------
    v16h aq[2][2];
    for (int mt = 0; mt < 2; ++mt)
        for (int kd = 0; kd < 2; ++kd)
            aq[mt][kd] = *(const v16h*)(Qb + (size_t)(nblk * 32 + mt * 16 + lm) * HDIM
                                           + kd * 32 + koff);
    for (int nt = 0; nt < 16; ++nt) {
        int keyb = w * 256 + nt * 16;
        v16h b0 = *(const v16h*)(Kb + (size_t)(keyb + lm) * HDIM + koff);
        v16h b1 = *(const v16h*)(Kb + (size_t)(keyb + lm) * HDIM + 32 + koff);
        for (int mt = 0; mt < 2; ++mt) {
            v8f acc = {};
            acc = wmma16(aq[mt][0], b0, acc);
            acc = wmma16(aq[mt][1], b1, acc);
            int col = keyb + lm;
            for (int r = 0; r < 8; ++r) {
                int row = mt * 16 + r + (lane >> 4) * 8;
                S[row * SROW + col] = acc[r] * ATTN_SCALE;
            }
        }
    }
    __syncthreads();

    // ---- Phase 2: softmax stats, exp written back in place --------------
    {
        int row = tid >> 3;
        int cl  = tid & 7;
        float mx = -1e30f;
        for (int c = cl; c < SEQN; c += 8)
            mx = fmaxf(mx, S[row * SROW + c]);
        red[row * 8 + cl] = mx;
        __syncthreads();
        if (tid < 32) {
            float m2 = red[tid * 8];
            for (int j = 1; j < 8; ++j) m2 = fmaxf(m2, red[tid * 8 + j]);
            rstat[tid] = m2;
        }
        __syncthreads();
        float rmax = rstat[row];
        float sum = 0.f;
        for (int c = cl; c < SEQN; c += 8) {
            float e = __expf(S[row * SROW + c] - rmax);
            S[row * SROW + c] = e;
            sum += e;
        }
        red[row * 8 + cl] = sum;
        __syncthreads();
        if (tid < 32) {
            float s2 = 0.f;
            for (int j = 0; j < 8; ++j) s2 += red[tid * 8 + j];
            rstat[tid] = 1.0f / s2;
        }
        __syncthreads();
    }

    // ---- Phase 3: O = P V (wave w -> rows m0..m0+15, dims d0..d0+15) ----
    {
        const int m0 = (w & 1) * 16;
        const int d0 = (w >> 1) * 16;
        const int srow = m0 + lm;
        v8f acc = {};
        for (int kk = 0; kk < SEQN; kk += 32) {
            const float* sp = &S[srow * SROW + kk + koff];
            v16h a;
            #pragma unroll
            for (int i = 0; i < 16; ++i) a[i] = (_Float16)sp[i];
            v16h b = *(const v16h*)(Vb + (size_t)(d0 + lm) * SEQN + kk + koff);
            acc = wmma16(a, b, acc);
        }
        const int bb = bh >> 4, hh = bh & 15;
        for (int r = 0; r < 8; ++r) {
            int rowl = m0 + r + (lane >> 4) * 8;
            int n = nblk * 32 + rowl;
            int d = d0 + lm;
            float val = acc[r] * rstat[rowl];
            oh[((size_t)bb * SEQN + n) * DIMC + hh * HDIM + d] = (_Float16)val;
        }
    }
}

// ---------------------------------------------------------------------------
// Host launcher
// ---------------------------------------------------------------------------
extern "C" void kernel_launch(void* const* d_in, const int* in_sizes, int n_in,
                              void* d_out, int out_size, void* d_ws, size_t ws_size,
                              hipStream_t stream) {
    (void)in_sizes; (void)n_in; (void)out_size; (void)ws_size;
    const float* x  = (const float*)d_in[0];   // (2,2048,1024)
    const float* Wq = (const float*)d_in[1];   // (1024,1024)
    const float* Wk = (const float*)d_in[2];
    const float* Wv = (const float*)d_in[3];
    const float* Wp = (const float*)d_in[4];
    const float* bp = (const float*)d_in[5];   // (1024,)
    float* out = (float*)d_out;                // (2,2048,1024) fp32

    const size_t NTOK = (size_t)NBATCH * SEQN;            // 4096
    const size_t XSZ  = NTOK * DIMC;                      // 4,194,304
    const size_t WSZ  = (size_t)DIMC * DIMC;              // 1,048,576

    char* ws = (char*)d_ws;
    size_t off = 0;
    auto carve = [&](size_t bytes) {
        void* p = ws + off;
        off += (bytes + 255) & ~(size_t)255;
        return p;
    };
    _Float16* xh  = (_Float16*)carve(XSZ * 2);
    _Float16* Wqh = (_Float16*)carve(WSZ * 2);
    _Float16* Wkh = (_Float16*)carve(WSZ * 2);
    _Float16* Wvh = (_Float16*)carve(WSZ * 2);
    _Float16* Wph = (_Float16*)carve(WSZ * 2);
    _Float16* qhB = (_Float16*)carve(XSZ * 2);   // (B,H,N,D)
    _Float16* khB = (_Float16*)carve(XSZ * 2);   // (B,H,N,D)
    _Float16* vTB = (_Float16*)carve(XSZ * 2);   // (B,H,D,N)
    _Float16* ohB = (_Float16*)carve(XSZ * 2);   // (B,N,C)

    // 1) fp32 -> f16 conversions
    {
        int n4 = (int)(XSZ / 4);
        cvt_f32_f16_kernel<<<(n4 + 255) / 256, 256, 0, stream>>>(x, xh, n4);
        int w4 = (int)(WSZ / 4);
        cvt_f32_f16_kernel<<<(w4 + 255) / 256, 256, 0, stream>>>(Wq, Wqh, w4);
        cvt_f32_f16_kernel<<<(w4 + 255) / 256, 256, 0, stream>>>(Wk, Wkh, w4);
        cvt_f32_f16_kernel<<<(w4 + 255) / 256, 256, 0, stream>>>(Wv, Wvh, w4);
        cvt_f32_f16_kernel<<<(w4 + 255) / 256, 256, 0, stream>>>(Wp, Wph, w4);
    }

    // 2) Q/K/V projections (TDM-staged WMMA GEMMs)
    gemm_wmma_kernel<0><<<512, 256, 0, stream>>>(xh, Wqh, qhB, nullptr, nullptr);
    gemm_wmma_kernel<0><<<512, 256, 0, stream>>>(xh, Wkh, khB, nullptr, nullptr);
    gemm_wmma_kernel<1><<<512, 256, 0, stream>>>(xh, Wvh, vTB, nullptr, nullptr);

    // 3) attention: 2*16 heads * 64 query-blocks = 2048 workgroups,
    //    dynamic LDS = score strip + reduction scratch (~258 KB of 320 KB)
    {
        size_t lds = (size_t)32 * SROW * 4 + 32 * 8 * 4 + 32 * 4;
        attn_kernel<<<NBATCH * NHEAD * (SEQN / 32), 256, lds, stream>>>(
            qhB, khB, vTB, ohB);
    }

    // 4) output projection with bias (fp32 epilogue)
    gemm_wmma_kernel<2><<<512, 256, 0, stream>>>(ohB, Wph, nullptr, out, bp);
}
